// InputToLeafSinkhorn_48670569398604
// MI455X (gfx1250) — matrix operand order repaired and checked by reading, not compiled, not run on hardware
//
#include <hip/hip_runtime.h>
#include <cstddef>

// ---------------------------------------------------------------------------
// InputToLeafSinkhorn on MI455X (gfx1250, wave32, WMMA)
//
// out = x @ P.T,  P = sinkhorn(exp(logits/T)), 20 iters row/col normalize.
// Sinkhorn is represented implicitly as P = diag(r) * A0 * diag(c):
//   per iter:  r = 1/(A0 c)   (row mat-vec),  c = 1/(A0^T r) (col mat-vec)
// GEMM: out[b,i] = r_i * sum_j (c_j * x[b,j]) * A0[i,j]
// computed with bf16x3 split-precision WMMA (hi*hi + hi*lo + lo*hi, f32 acc).
// ---------------------------------------------------------------------------

#define NLEAF 1024
#define NIN   1024
#define BATCH 32768
#define NITER 20

typedef __attribute__((ext_vector_type(16))) __bf16        v16bf;
typedef __attribute__((ext_vector_type(8)))  float         v8f;
typedef __attribute__((ext_vector_type(8)))  unsigned int  v8u;

static __device__ __forceinline__ unsigned short f2bf(float f) {
  unsigned int u = __builtin_bit_cast(unsigned int, f);
  u += 0x7FFFu + ((u >> 16) & 1u);          // round-to-nearest-even
  return (unsigned short)(u >> 16);
}
static __device__ __forceinline__ float bf2f(unsigned short h) {
  unsigned int u = ((unsigned int)h) << 16;
  return __builtin_bit_cast(float, u);
}

// ---------------- A0 = exp(logits / T) ----------------
__global__ void k_exp(const float* __restrict__ logits, float* __restrict__ A0) {
  const float invT = 1.0f / (0.05f + 1e-06f);
  int i = blockIdx.x * 256 + threadIdx.x;   // one float4 per thread
  float4 v = ((const float4*)logits)[i];
  float4 o;
  o.x = expf(v.x * invT);
  o.y = expf(v.y * invT);
  o.z = expf(v.z * invT);
  o.w = expf(v.w * invT);
  ((float4*)A0)[i] = o;
}

__global__ void k_ones(float* __restrict__ c) {
  c[blockIdx.x * 256 + threadIdx.x] = 1.0f;
}

// ---------------- r = 1 / (A0 @ c), one block per row ----------------
__global__ void k_rowmv(const float* __restrict__ A0, const float* __restrict__ c,
                        float* __restrict__ r) {
  __shared__ float red[256];
  const int row = blockIdx.x;
  const int t = threadIdx.x;
  const float* ap = A0 + (size_t)row * NIN;
  float acc = 0.0f;
#pragma unroll
  for (int j = t; j < NIN; j += 256) acc += ap[j] * c[j];
  red[t] = acc;
  __syncthreads();
#pragma unroll
  for (int s = 128; s > 0; s >>= 1) {
    if (t < s) red[t] += red[t + s];
    __syncthreads();
  }
  if (t == 0) r[row] = 1.0f / red[0];
}

// ---------------- c = 1 / (A0^T @ r), 64 columns per block ----------------
__global__ void k_colmv(const float* __restrict__ A0, const float* __restrict__ r,
                        float* __restrict__ c) {
  __shared__ float rs[NLEAF];
  __shared__ float red[256];
  const int t = threadIdx.x;
  for (int i = t; i < NLEAF; i += 256) rs[i] = r[i];
  __syncthreads();
  const int col = blockIdx.x * 64 + (t & 63);
  const int ph  = t >> 6;                   // 4 row-phases
  float acc = 0.0f;
  for (int i = ph; i < NLEAF; i += 4) acc += A0[(size_t)i * NIN + col] * rs[i];
  red[t] = acc;
  __syncthreads();
  if (ph == 0) {
    int cl = t & 63;
    c[col] = 1.0f / (red[cl] + red[64 + cl] + red[128 + cl] + red[192 + cl]);
  }
}

// ---------------- GEMM: out = (x*diag(c)) @ A0^T * diag(r) ----------------
// Block tile 128x128, BK=32, 256 threads = 8 wave32s (4 M x 2 N),
// wave tile 32x64 = 2x4 accumulators of 16x16.
// bf16x3: acc += Xhi*Bhi + Xhi*Blo + Xlo*Bhi  (v_wmma_f32_16x16x32_bf16)
#define BM 128
#define BN 128
#define BK 32
#define KSTEPS (NIN / BK)
#define LDST 48   // padded LDS row stride (bf16 elems): 96B, 16B-aligned reads

__global__ __launch_bounds__(256) void k_gemm(
    const float* __restrict__ X, const float* __restrict__ A0,
    const float* __restrict__ rsc, const float* __restrict__ csc,
    float* __restrict__ out) {
  __shared__ alignas(16) unsigned short Xhi[BM * LDST];
  __shared__ alignas(16) unsigned short Xlo[BM * LDST];
  __shared__ alignas(16) unsigned short Bhi[BN * LDST];
  __shared__ alignas(16) unsigned short Blo[BN * LDST];

  const int t     = threadIdx.x;
  const int lane  = t & 31;
  const int w     = t >> 5;
  const int waveM = w & 3;        // 0..3  -> 32-row slab
  const int waveN = w >> 2;       // 0..1  -> 64-col slab
  const int mBase = blockIdx.y * BM;
  const int nBase = blockIdx.x * BN;

  // cooperative global-load mapping: 1024 float4 per tile, 4 per thread
  float4 xr[4], br[4];
#pragma unroll
  for (int u = 0; u < 4; ++u) {
    const int id  = t + u * 256;
    const int row = id >> 3, c4 = id & 7;
    xr[u] = *(const float4*)(X  + (size_t)(mBase + row) * NIN + c4 * 4);
    br[u] = *(const float4*)(A0 + (size_t)(nBase + row) * NIN + c4 * 4);
  }

  v8f acc[2][4];
#pragma unroll
  for (int mi = 0; mi < 2; ++mi)
#pragma unroll
    for (int ni = 0; ni < 4; ++ni)
      acc[mi][ni] = (v8f){0.f, 0.f, 0.f, 0.f, 0.f, 0.f, 0.f, 0.f};

#pragma unroll 1
  for (int kb = 0; kb < KSTEPS; ++kb) {
    // ---- split current fp32 tiles into bf16 hi/lo in LDS ----
#pragma unroll
    for (int u = 0; u < 4; ++u) {
      const int id  = t + u * 256;
      const int row = id >> 3, c4 = id & 7;
      const float4 cv = *(const float4*)(csc + kb * BK + c4 * 4);
      const float fx[4] = {xr[u].x * cv.x, xr[u].y * cv.y,
                           xr[u].z * cv.z, xr[u].w * cv.w};
      const float fb[4] = {br[u].x, br[u].y, br[u].z, br[u].w};
      unsigned short xh[4], xl[4], bh[4], bl[4];
#pragma unroll
      for (int e = 0; e < 4; ++e) {
        xh[e] = f2bf(fx[e]); xl[e] = f2bf(fx[e] - bf2f(xh[e]));
        bh[e] = f2bf(fb[e]); bl[e] = f2bf(fb[e] - bf2f(bh[e]));
      }
      const int o = row * LDST + c4 * 4;    // 8B-aligned (c4*4*2 bytes)
      *(uint2*)&Xhi[o] = make_uint2(xh[0] | ((unsigned)xh[1] << 16),
                                    xh[2] | ((unsigned)xh[3] << 16));
      *(uint2*)&Xlo[o] = make_uint2(xl[0] | ((unsigned)xl[1] << 16),
                                    xl[2] | ((unsigned)xl[3] << 16));
      *(uint2*)&Bhi[o] = make_uint2(bh[0] | ((unsigned)bh[1] << 16),
                                    bh[2] | ((unsigned)bh[3] << 16));
      *(uint2*)&Blo[o] = make_uint2(bl[0] | ((unsigned)bl[1] << 16),
                                    bl[2] | ((unsigned)bl[3] << 16));
    }
    __syncthreads();

    // ---- prefetch next K-step into registers (hides HBM latency) ----
    if (kb + 1 < KSTEPS) {
      const int k0 = (kb + 1) * BK;
#pragma unroll
      for (int u = 0; u < 4; ++u) {
        const int id  = t + u * 256;
        const int row = id >> 3, c4 = id & 7;
        xr[u] = *(const float4*)(X  + (size_t)(mBase + row) * NIN + k0 + c4 * 4);
        br[u] = *(const float4*)(A0 + (size_t)(nBase + row) * NIN + k0 + c4 * 4);
      }
    }

    // ---- A fragments (16x32 bf16): lane<16 holds K0..7 & K16..23 ----
    v16bf ahi[2], alo[2];
#pragma unroll
    for (int mi = 0; mi < 2; ++mi) {
      const int m    = waveM * 32 + mi * 16 + (lane & 15);
      const int koff = (lane >> 4) * 8;
      const uint4 h0 = *(const uint4*)&Xhi[m * LDST + koff];
      const uint4 h1 = *(const uint4*)&Xhi[m * LDST + 16 + koff];
      const uint4 l0 = *(const uint4*)&Xlo[m * LDST + koff];
      const uint4 l1 = *(const uint4*)&Xlo[m * LDST + 16 + koff];
      v8u qh, ql;
      qh[0]=h0.x; qh[1]=h0.y; qh[2]=h0.z; qh[3]=h0.w;
      qh[4]=h1.x; qh[5]=h1.y; qh[6]=h1.z; qh[7]=h1.w;
      ql[0]=l0.x; ql[1]=l0.y; ql[2]=l0.z; ql[3]=l0.w;
      ql[4]=l1.x; ql[5]=l1.y; ql[6]=l1.z; ql[7]=l1.w;
      ahi[mi] = __builtin_bit_cast(v16bf, qh);
      alo[mi] = __builtin_bit_cast(v16bf, ql);
    }

    // ---- B fragments (32x16 bf16): lane<16 holds K0..7 & K16..23 ----
#pragma unroll
    for (int ni = 0; ni < 4; ++ni) {
      const int n    = waveN * 64 + ni * 16 + (lane & 15);
      const int koff = (lane >> 4) * 8;
      const uint4 h0 = *(const uint4*)&Bhi[n * LDST + koff];
      const uint4 h1 = *(const uint4*)&Bhi[n * LDST + koff + 16];
      const uint4 l0 = *(const uint4*)&Blo[n * LDST + koff];
      const uint4 l1 = *(const uint4*)&Blo[n * LDST + koff + 16];
      v8u qh, ql;
      qh[0]=h0.x; qh[1]=h0.y; qh[2]=h0.z; qh[3]=h0.w;
      qh[4]=h1.x; qh[5]=h1.y; qh[6]=h1.z; qh[7]=h1.w;
      ql[0]=l0.x; ql[1]=l0.y; ql[2]=l0.z; ql[3]=l0.w;
      ql[4]=l1.x; ql[5]=l1.y; ql[6]=l1.z; ql[7]=l1.w;
      const v16bf bhi = __builtin_bit_cast(v16bf, qh);
      const v16bf blo = __builtin_bit_cast(v16bf, ql);
#pragma unroll
      for (int mi = 0; mi < 2; ++mi) {
        acc[mi][ni] = __builtin_amdgcn_wmma_f32_16x16x32_bf16(
            false, ahi[mi], false, bhi, (short)0, acc[mi][ni], false, false);
        acc[mi][ni] = __builtin_amdgcn_wmma_f32_16x16x32_bf16(
            false, ahi[mi], false, blo, (short)0, acc[mi][ni], false, false);
        acc[mi][ni] = __builtin_amdgcn_wmma_f32_16x16x32_bf16(
            false, alo[mi], false, bhi, (short)0, acc[mi][ni], false, false);
      }
    }
    __syncthreads();
  }

  // ---- epilogue: scale output columns by r and store fp32 ----
#pragma unroll
  for (int ni = 0; ni < 4; ++ni) {
    const int n = nBase + waveN * 64 + ni * 16 + (lane & 15);
    const float rn = rsc[n];
#pragma unroll
    for (int mi = 0; mi < 2; ++mi) {
#pragma unroll
      for (int d = 0; d < 8; ++d) {
        const int rowo = mBase + waveM * 32 + mi * 16 + d + (lane >> 4) * 8;
        out[(size_t)rowo * NLEAF + n] = acc[mi][ni][d] * rn;
      }
    }
  }
}

extern "C" void kernel_launch(void* const* d_in, const int* in_sizes, int n_in,
                              void* d_out, int out_size, void* d_ws, size_t ws_size,
                              hipStream_t stream) {
  const float* x      = (const float*)d_in[0];   // [32768, 1024]
  const float* logits = (const float*)d_in[1];   // [1024, 1024]
  float* out = (float*)d_out;                    // [32768, 1024]

  float* A0 = (float*)d_ws;                      // 1024*1024 fp32 = 4 MB
  float* r  = A0 + (size_t)NLEAF * NIN;          // 1024 fp32
  float* c  = r + NLEAF;                         // 1024 fp32

  k_exp <<<(NLEAF * NIN) / (256 * 4), 256, 0, stream>>>(logits, A0);
  k_ones<<<NIN / 256, 256, 0, stream>>>(c);
  for (int it = 0; it < NITER; ++it) {
    k_rowmv<<<NLEAF, 256, 0, stream>>>(A0, c, r);
    k_colmv<<<NIN / 64, 256, 0, stream>>>(A0, r, c);
  }
  k_gemm<<<dim3(NLEAF / BN, BATCH / BM), 256, 0, stream>>>(x, A0, r, c, out);
}
